// CFODE_19516331393404
// MI455X (gfx1250) — matrix operand order repaired and checked by reading, not compiled, not run on hardware
//
#include <hip/hip_runtime.h>
#include <math.h>

// ---------------------------------------------------------------- WMMA types
typedef __attribute__((ext_vector_type(16))) __bf16 bf16x16;
typedef __attribute__((ext_vector_type(8)))  float  f32x8;

union FragU { bf16x16 v; uint4 q[2]; };

#define WMMA_BF16(A, B, C) \
  __builtin_amdgcn_wmma_f32_16x16x32_bf16(false, (A), false, (B), (short)0, (C), false, false)

// A fragment: 16x32 bf16 tile of row-major [.., ld] matrix, base at (m0,k0).
// ISA layout: lane<16 -> row=lane, K chunks {0..7},{16..23};
//             lane>=16 -> row=lane-16, K chunks {8..15},{24..31}.
__device__ __forceinline__ bf16x16 load_frag_a(const __bf16* p, int ld) {
  int lane = threadIdx.x & 31;
  int m  = lane & 15;
  int hi = lane >> 4;
  const char* base = (const char*)(p + (size_t)m * ld + hi * 8);
  FragU f;
  f.q[0] = *(const uint4*)(base);
  f.q[1] = *(const uint4*)(base + 32);
  return f.v;
}

// B fragment: 32x16 bf16 (B[k][n] = W[n][k]) from row-major W, base at (n0,k0).
// ISA layout: lane<16 -> col n=lane, K chunks {0..7},{16..23}; lane>=16 -> col n=lane-16, K chunks {8..15},{24..31}.
__device__ __forceinline__ bf16x16 load_frag_b(const __bf16* p, int ld) {
  int lane = threadIdx.x & 31;
  int n  = lane & 15;
  int hi = lane >> 4;
  const char* base = (const char*)(p + (size_t)n * ld + hi * 8);
  FragU f;
  f.q[0] = *(const uint4*)(base);
  f.q[1] = *(const uint4*)(base + 32);
  return f.v;
}

__device__ __forceinline__ float sigmoidf_(float x) { return 1.f / (1.f + __expf(-x)); }

// ---------------------------------------------------------------- stats/norm
// per-batch temporal stats; builds enc_in[b][l][28] and tr[b][p][8]; saves om/os.
__global__ void stats_kernel(const float* __restrict__ cov, const float* __restrict__ tre,
                             const float* __restrict__ outh, const float* __restrict__ trt,
                             float* __restrict__ enc_in, float* __restrict__ tr,
                             float* __restrict__ om, float* __restrict__ os_) {
  int b = blockIdx.x;
  int c = threadIdx.x;           // 0..31, 28 used
  if (c >= 28) return;
  const float* src; int nch, ch;
  if (c < 16)      { src = cov;  nch = 16; ch = c; }
  else if (c < 24) { src = tre;  nch = 8;  ch = c - 16; }
  else             { src = outh; nch = 4;  ch = c - 24; }
  float s = 0.f, ss = 0.f;
  for (int l = 0; l < 64; ++l) {
    float v = src[((size_t)b * 64 + l) * nch + ch];
    s += v; ss += v * v;
  }
  float m = s / 64.f;
  float var = ss / 64.f - m * m;
  float sd = sqrtf(fmaxf(var, 0.f));
  float inv = 1.f / (sd + 1e-5f);
  for (int l = 0; l < 64; ++l) {
    float v = src[((size_t)b * 64 + l) * nch + ch];
    enc_in[((size_t)b * 64 + l) * 28 + c] = (v - m) * inv;
  }
  if (c >= 16 && c < 24) {       // normalize future treatments with history stats
    for (int p = 0; p < 8; ++p) {
      float v = trt[((size_t)b * 8 + p) * 8 + ch];
      tr[((size_t)b * 8 + p) * 8 + ch] = (v - m) * inv;
    }
  }
  if (c >= 24) { om[b * 4 + ch] = m; os_[b * 4 + ch] = sd; }
}

// ---------------------------------------------------------------- gi precompute
// gi[bl][g] = x[bl][:] . Wih[g][:] + bih[g]  (K = 28 or 8; tiny, scalar is fine)
__global__ void gi_kernel(const float* __restrict__ x, const float* __restrict__ Wih,
                          const float* __restrict__ bih, float* __restrict__ gi,
                          int L, int K) {
  int idx = blockIdx.x * blockDim.x + threadIdx.x;
  int total = 128 * L * 384;
  if (idx >= total) return;
  int g = idx % 384, bl = idx / 384;
  const float* xp = x + (size_t)bl * K;
  const float* wp = Wih + (size_t)g * K;
  float s = bih[g];
  for (int k = 0; k < K; ++k) s += xp[k] * wp[k];
  gi[idx] = s;
}

__global__ void cvt_bf16_kernel(const float* __restrict__ src, __bf16* __restrict__ dst, int n) {
  int i = blockIdx.x * blockDim.x + threadIdx.x;
  if (i < n) dst[i] = (__bf16)src[i];
}

// ---------------------------------------------------------------- fused bi-GRU
// 4 persistent blocks: {enc_f, enc_b, dec_f, dec_b}. Whh(bf16,96KB) + h(f32,64KB)
// + h(bf16,32KB) resident in LDS; per step a 128x384x128 WMMA GEMM + gates.
__global__ void __launch_bounds__(512, 1)
gru_kernel(const float* gi_ef, const float* gi_eb, const float* gi_df, const float* gi_db,
           const float* whh_ef, const float* whh_eb, const float* whh_df, const float* whh_db,
           const float* bhh_ef, const float* bhh_eb, const float* bhh_df, const float* bhh_db,
           float* __restrict__ hT) {
  extern __shared__ char smem[];
  __bf16* sW  = (__bf16*)smem;                       // 384*128 bf16 = 96KB
  float*  sHf = (float*)(smem + 98304);              // 128*128 f32  = 64KB
  __bf16* sHb = (__bf16*)(smem + 98304 + 65536);     // 128*128 bf16 = 32KB
  float*  sB  = (float*)(smem + 98304 + 65536 + 32768); // 384 f32

  int blk = blockIdx.x;
  const float* gi  = (blk == 0) ? gi_ef : (blk == 1) ? gi_eb : (blk == 2) ? gi_df : gi_db;
  const float* whh = (blk == 0) ? whh_ef : (blk == 1) ? whh_eb : (blk == 2) ? whh_df : whh_db;
  const float* bhh = (blk == 0) ? bhh_ef : (blk == 1) ? bhh_eb : (blk == 2) ? bhh_df : bhh_db;
  int  L   = (blk < 2) ? 64 : 8;
  bool rev = (blk & 1);

  int tid = threadIdx.x, wave = tid >> 5, lane = tid & 31;
  for (int i = tid; i < 384 * 128; i += 512) sW[i] = (__bf16)whh[i];
  for (int i = tid; i < 128 * 128; i += 512) { sHf[i] = 0.f; sHb[i] = (__bf16)0.f; }
  for (int i = tid; i < 384; i += 512) sB[i] = bhh[i];
  __syncthreads();

  for (int t = 0; t < L; ++t) {
    int ti = rev ? (L - 1 - t) : t;
    f32x8 acc[4][3];
    // ---- GEMM phase: gh = h @ Whh^T, 64 (m,nh) tile-pairs over 16 waves
#pragma unroll
    for (int p = 0; p < 4; ++p) {
      int pair = wave + 16 * p;
      int m = pair & 7, nh = pair >> 3;
      f32x8 a0 = {}, a1 = {}, a2 = {};
#pragma unroll
      for (int kb = 0; kb < 4; ++kb) {
        int k0 = kb * 32;
        bf16x16 af = load_frag_a(sHb + (size_t)(m * 16) * 128 + k0, 128);
        bf16x16 b0 = load_frag_b(sW + (size_t)(0 * 128 + nh * 16) * 128 + k0, 128);
        bf16x16 b1 = load_frag_b(sW + (size_t)(1 * 128 + nh * 16) * 128 + k0, 128);
        bf16x16 b2 = load_frag_b(sW + (size_t)(2 * 128 + nh * 16) * 128 + k0, 128);
        a0 = WMMA_BF16(af, b0, a0);
        a1 = WMMA_BF16(af, b1, a1);
        a2 = WMMA_BF16(af, b2, a2);
      }
      acc[p][0] = a0; acc[p][1] = a1; acc[p][2] = a2;
    }
    __syncthreads();   // all reads of sHb done before updates
    // ---- gate phase (each wave owns its (m,nh) blocks exclusively)
#pragma unroll
    for (int p = 0; p < 4; ++p) {
      int pair = wave + 16 * p;
      int m = pair & 7, nh = pair >> 3;
      int col = nh * 16 + (lane & 15);
      int mb  = m * 16 + ((lane >> 4) << 3);
      float hrb = sB[col], hzb = sB[128 + col], hnb = sB[256 + col];
#pragma unroll
      for (int v = 0; v < 8; ++v) {
        int row = mb + v;   // batch index
        size_t gib = ((size_t)row * L + ti) * 384;
        float ir = gi[gib + col], iz = gi[gib + 128 + col], inn = gi[gib + 256 + col];
        float hr = acc[p][0][v] + hrb;
        float hz = acc[p][1][v] + hzb;
        float hn = acc[p][2][v] + hnb;
        float r  = sigmoidf_(ir + hr);
        float z  = sigmoidf_(iz + hz);
        float nv = tanhf(inn + r * hn);
        float hold = sHf[row * 128 + col];
        float hnew = (1.f - z) * nv + z * hold;
        sHf[row * 128 + col] = hnew;
        sHb[row * 128 + col] = (__bf16)hnew;
      }
    }
    __syncthreads();
  }
  float* out = hT + (size_t)blk * 128 * 128;
  for (int i = tid; i < 128 * 128; i += 512) out[i] = sHf[i];
}

// x0[b][0:128]=0.5*(hT0+hT1); x0[b][128:256]=0.5*(hT2+hT3)
__global__ void combine_kernel(const float* __restrict__ hT, float* __restrict__ x0) {
  int idx = blockIdx.x * blockDim.x + threadIdx.x;
  if (idx >= 128 * 256) return;
  int b = idx >> 8, c = idx & 255;
  float v;
  if (c < 128) v = 0.5f * (hT[b * 128 + c] + hT[16384 + b * 128 + c]);
  else         v = 0.5f * (hT[32768 + b * 128 + (c - 128)] + hT[49152 + b * 128 + (c - 128)]);
  x0[idx] = v;
}

// ---------------------------------------------------------------- SDE scan
// 8 persistent WGs, each owns 16 batch rows for the whole 140-step scan.
// Per step: 3-layer tanh+LN MLP drift (WMMA bf16, weights L2-resident),
// Girsanov term, Euler-Maruyama update; snapshot every 20 steps.
__device__ __forceinline__ void mlp_layer(const __bf16* inB, int K, const __bf16* W, int N,
                                          const float* __restrict__ bias,
                                          const float* __restrict__ gain,
                                          const float* __restrict__ beta,
                                          float* outF, __bf16* outB) {
  int tid = threadIdx.x, wave = tid >> 5, lane = tid & 31;
  int ntiles = N >> 4;
  for (int nt = wave; nt < ntiles; nt += 16) {
    f32x8 acc = {};
    for (int k0 = 0; k0 < K; k0 += 32) {
      bf16x16 a = load_frag_a(inB + k0, K);
      bf16x16 b = load_frag_b(W + (size_t)(nt * 16) * K + k0, K);
      acc = WMMA_BF16(a, b, acc);
    }
    int n  = nt * 16 + (lane & 15);
    int mb = (lane >> 4) << 3;
    float bn = bias[n];
#pragma unroll
    for (int v = 0; v < 8; ++v) outF[(mb + v) * N + n] = tanhf(acc[v] + bn);
  }
  __syncthreads();
  // LayerNorm: wave w normalizes row w (16 rows)
  {
    int r = wave;
    int per = N >> 5;
    float s = 0.f, ss = 0.f;
    for (int j = 0; j < per; ++j) {
      float v = outF[r * N + lane + (j << 5)];
      s += v; ss += v * v;
    }
    for (int o = 16; o; o >>= 1) { s += __shfl_xor(s, o, 32); ss += __shfl_xor(ss, o, 32); }
    float mean = s / (float)N;
    float inv  = rsqrtf(ss / (float)N - mean * mean + 1e-5f);
    for (int j = 0; j < per; ++j) {
      int c = lane + (j << 5);
      float v = (outF[r * N + c] - mean) * inv * gain[c] + beta[c];
      outF[r * N + c] = v;
      if (outB) outB[r * N + c] = (__bf16)v;
    }
  }
  __syncthreads();
}

__global__ void __launch_bounds__(512, 1)
sde_kernel(const float* __restrict__ x0,
           const __bf16* __restrict__ W0, const __bf16* __restrict__ W1, const __bf16* __restrict__ W2,
           const float* b0, const float* g0, const float* be0,
           const float* b1, const float* g1, const float* be1,
           const float* b2, const float* g2, const float* be2,
           const float* __restrict__ noise, float* __restrict__ ys) {
  extern __shared__ char smem[];
  float*  sY   = (float*)(smem);                 // 16*256 f32
  __bf16* sYb  = (__bf16*)(smem + 16384);        // 16*256 bf16
  float*  sA1  = (float*)(smem + 24576);         // 16*512 f32
  __bf16* sA1b = (__bf16*)(smem + 57344);        // 16*512 bf16
  float*  sA2  = (float*)(smem + 73728);         // 16*512 f32
  __bf16* sA2b = (__bf16*)(smem + 106496);       // 16*512 bf16
  float*  sF   = (float*)(smem + 122880);        // 16*256 f32
  float*  sLq  = (float*)(smem + 139264);        // 16 f32

  int wg = blockIdx.x;               // 0..7
  int brow0 = wg * 16;
  int tid = threadIdx.x, wave = tid >> 5, lane = tid & 31;

  for (int i = tid; i < 16 * 256; i += 512) {
    int r = i >> 8, c = i & 255;
    float v = x0[(size_t)(brow0 + r) * 256 + c];
    sY[i] = v; sYb[i] = (__bf16)v;
  }
  if (tid < 16) sLq[tid] = 0.f;
  __syncthreads();
  // snapshot 0 = aug0
  for (int i = tid; i < 16 * 257; i += 512) {
    int r = i / 257, c = i % 257;
    ys[((size_t)0 * 128 + brow0 + r) * 257 + c] = (c < 256) ? sY[r * 256 + c] : 0.f;
  }
  const float sigsq = 0.5f * 0.22360679774997896f;  // SIGMA * sqrt(DT)
  for (int t = 0; t < 140; ++t) {
    mlp_layer(sYb, 256, W0, 512, b0, g0, be0, sA1, sA1b);
    mlp_layer(sA1b, 512, W1, 512, b1, g1, be1, sA2, sA2b);
    mlp_layer(sA2b, 512, W2, 256, b2, g2, be2, sF, (__bf16*)0);
    // Euler-Maruyama + Girsanov: wave r handles batch row r
    {
      int r = wave;
      int brow = brow0 + r;
      float s = 0.f;
#pragma unroll
      for (int j = 0; j < 8; ++j) {
        int c = lane + (j << 5);
        float yv = sY[r * 256 + c];
        float fv = sF[r * 256 + c];
        float u  = 2.0f * (fv + yv);   // (f - THETA*(MU - y)) / SIGMA
        s += u * u;
        float nz = noise[((size_t)t * 128 + brow) * 256 + c];
        float yn = yv + fv * 0.05f + sigsq * nz;
        sY[r * 256 + c] = yn;
        sYb[r * 256 + c] = (__bf16)yn;
      }
      for (int o = 16; o; o >>= 1) s += __shfl_xor(s, o, 32);
      if (lane == 0) sLq[r] += 0.5f * s * 0.05f;
    }
    __syncthreads();
    if (((t + 1) % 20) == 0) {
      int snap = (t + 1) / 20;
      for (int i = tid; i < 16 * 257; i += 512) {
        int r = i / 257, c = i % 257;
        ys[((size_t)snap * 128 + brow0 + r) * 257 + c] = (c < 256) ? sY[r * 256 + c] : sLq[r];
      }
      __syncthreads();
    }
  }
}

// ---------------------------------------------------------------- output head
__global__ void out_kernel(const float* __restrict__ ys, const float* __restrict__ outW,
                           const float* __restrict__ outb, const float* __restrict__ lnw,
                           const float* __restrict__ lnb, const float* __restrict__ om,
                           const float* __restrict__ os_, float* __restrict__ dout) {
  int b = blockIdx.x;
  __shared__ float dec[8][8];     // [p][o]
  int tid = threadIdx.x;          // 128
  if (tid < 64) {
    int p = tid >> 3, o = tid & 7;
    const float* yv = ys + ((size_t)p * 128 + b) * 257;
    const float* w  = outW + o * 128;
    float s = outb[o];
    for (int k = 0; k < 128; ++k) s += yv[k] * w[k];
    dec[p][o] = s;
  }
  __syncthreads();
  if (tid < 32) {                 // mu
    int p = tid >> 2, o = tid & 3;
    dout[((size_t)b * 8 + p) * 4 + o] = dec[p][o] * (os_[b * 4 + o] + 1e-5f) + om[b * 4 + o];
  } else if (tid < 64) {          // var (LN over the 8 horizon steps, then sigmoid)
    int p = (tid - 32) >> 2, o = (tid - 32) & 3;
    float m = 0.f, ss = 0.f;
    for (int q = 0; q < 8; ++q) { float v = dec[q][4 + o]; m += v; ss += v * v; }
    m *= 0.125f;
    float vv = ss * 0.125f - m * m;
    float x = (dec[p][4 + o] - m) * rsqrtf(vv + 1e-5f) * lnw[o] + lnb[o];
    dout[4096 + ((size_t)b * 8 + p) * 4 + o] = sigmoidf_(x);
  }
  for (int i = tid; i < 129 * 8; i += blockDim.x) {   // logqp[b][d][p]
    int d = i >> 3, p = i & 7;
    dout[8192 + ((size_t)b * 129 + d) * 8 + p] = ys[((size_t)p * 128 + b) * 257 + 128 + d];
  }
}

// ---------------------------------------------------------------- launch
extern "C" void kernel_launch(void* const* d_in, const int* in_sizes, int n_in,
                              void* d_out, int out_size, void* d_ws, size_t ws_size,
                              hipStream_t stream) {
  const float* cov  = (const float*)d_in[0];
  const float* tre  = (const float*)d_in[1];
  const float* outh = (const float*)d_in[2];
  const float* trt  = (const float*)d_in[3];
  const float* eWihF = (const float*)d_in[5],  *eWhhF = (const float*)d_in[6];
  const float* ebihF = (const float*)d_in[7],  *ebhhF = (const float*)d_in[8];
  const float* eWihB = (const float*)d_in[9],  *eWhhB = (const float*)d_in[10];
  const float* ebihB = (const float*)d_in[11], *ebhhB = (const float*)d_in[12];
  const float* dWihF = (const float*)d_in[13], *dWhhF = (const float*)d_in[14];
  const float* dbihF = (const float*)d_in[15], *dbhhF = (const float*)d_in[16];
  const float* dWihB = (const float*)d_in[17], *dWhhB = (const float*)d_in[18];
  const float* dbihB = (const float*)d_in[19], *dbhhB = (const float*)d_in[20];
  const float* mW0 = (const float*)d_in[21], *mb0 = (const float*)d_in[22];
  const float* mg0 = (const float*)d_in[23], *mbe0 = (const float*)d_in[24];
  const float* mW1 = (const float*)d_in[25], *mb1 = (const float*)d_in[26];
  const float* mg1 = (const float*)d_in[27], *mbe1 = (const float*)d_in[28];
  const float* mW2 = (const float*)d_in[29], *mb2 = (const float*)d_in[30];
  const float* mg2 = (const float*)d_in[31], *mbe2 = (const float*)d_in[32];
  const float* outW = (const float*)d_in[33], *outb = (const float*)d_in[34];
  const float* lnw = (const float*)d_in[35],  *lnb = (const float*)d_in[36];
  const float* noise = (const float*)d_in[37];

  // workspace partition (256B aligned)
  char* ws = (char*)d_ws;
  size_t off = 0;
  auto take = [&](size_t bytes) { size_t r = off; off = (off + bytes + 255) & ~(size_t)255; return r; };
  float* enc_in = (float*)(ws + take(128ull * 64 * 28 * 4));
  float* tr     = (float*)(ws + take(128ull * 8 * 8 * 4));
  float* om     = (float*)(ws + take(128ull * 4 * 4));
  float* os_    = (float*)(ws + take(128ull * 4 * 4));
  float* gi_ef  = (float*)(ws + take(128ull * 64 * 384 * 4));
  float* gi_eb  = (float*)(ws + take(128ull * 64 * 384 * 4));
  float* gi_df  = (float*)(ws + take(128ull * 8 * 384 * 4));
  float* gi_db  = (float*)(ws + take(128ull * 8 * 384 * 4));
  float* hT     = (float*)(ws + take(4ull * 128 * 128 * 4));
  float* x0     = (float*)(ws + take(128ull * 256 * 4));
  float* ysbuf  = (float*)(ws + take(8ull * 128 * 257 * 4));
  __bf16* W0b   = (__bf16*)(ws + take(512ull * 256 * 2));
  __bf16* W1b   = (__bf16*)(ws + take(512ull * 512 * 2));
  __bf16* W2b   = (__bf16*)(ws + take(256ull * 512 * 2));

  // 1) stats + normalization
  stats_kernel<<<128, 32, 0, stream>>>(cov, tre, outh, trt, enc_in, tr, om, os_);
  // 2) gi precompute (4 weight sets)
  {
    int totE = 128 * 64 * 384, totD = 128 * 8 * 384;
    gi_kernel<<<(totE + 255) / 256, 256, 0, stream>>>(enc_in, eWihF, ebihF, gi_ef, 64, 28);
    gi_kernel<<<(totE + 255) / 256, 256, 0, stream>>>(enc_in, eWihB, ebihB, gi_eb, 64, 28);
    gi_kernel<<<(totD + 255) / 256, 256, 0, stream>>>(tr, dWihF, dbihF, gi_df, 8, 8);
    gi_kernel<<<(totD + 255) / 256, 256, 0, stream>>>(tr, dWihB, dbihB, gi_db, 8, 8);
  }
  // 3) MLP weights -> bf16 (L2-resident thereafter)
  cvt_bf16_kernel<<<(512 * 256 + 255) / 256, 256, 0, stream>>>(mW0, W0b, 512 * 256);
  cvt_bf16_kernel<<<(512 * 512 + 255) / 256, 256, 0, stream>>>(mW1, W1b, 512 * 512);
  cvt_bf16_kernel<<<(256 * 512 + 255) / 256, 256, 0, stream>>>(mW2, W2b, 256 * 512);
  // 4) fused bi-GRUs: 4 persistent WGPs, Whh in LDS, WMMA per step
  gru_kernel<<<4, 512, 198144, stream>>>(gi_ef, gi_eb, gi_df, gi_db,
                                         eWhhF, eWhhB, dWhhF, dWhhB,
                                         ebhhF, ebhhB, dbhhF, dbhhB, hT);
  // 5) x0 = [enc_h ; dec_h]
  combine_kernel<<<(128 * 256 + 255) / 256, 256, 0, stream>>>(hT, x0);
  // 6) latent SDE scan: 8 persistent WGs, 140 steps, WMMA drift MLP
  sde_kernel<<<8, 512, 139328, stream>>>(x0, W0b, W1b, W2b,
                                         mb0, mg0, mbe0, mb1, mg1, mbe1, mb2, mg2, mbe2,
                                         noise, ysbuf);
  // 7) output head
  out_kernel<<<128, 128, 0, stream>>>(ysbuf, outW, outb, lnw, lnb, om, os_, (float*)d_out);
}
